// MDTGCN_15899968930458
// MI455X (gfx1250) — matrix operand order repaired and checked by reading, not compiled, not run on hardware
//
#include <hip/hip_runtime.h>

// ---------------------------------------------------------------------------
// MDT-GCN forward for MI455X (gfx1250, wave32).
// Heavy math on v_wmma_f32_16x16x32_bf16 (f32 accumulation), operands staged
// bf16. All GEMM operands are staged k-contiguous ([M][K] / [N][K]) with
// K-strides padded to multiples of 32 elements, so every fragment is two
// 16B-aligned global_load_b128 with immediate offsets: no strided gathers,
// no address spills, no K tail. A-side pads are zero-filled (0*garbage = 0).
// Bilinear 3x3->92x124 upsample is fused into the encoder's first conv.
// Requires ws_size >= ~600 MB.
// ---------------------------------------------------------------------------

typedef __bf16 bf16;
typedef bf16  v16bf __attribute__((ext_vector_type(16)));
typedef bf16  v8bf  __attribute__((ext_vector_type(8)));
typedef float v8f   __attribute__((ext_vector_type(8)));

enum { GF_RELU = 4, GF_OUTBF = 8, GF_BIAS = 16, GF_CTRANS = 32 };

static __device__ __forceinline__ int imin(int a, int b) { return a < b ? a : b; }
static __device__ __forceinline__ int imax(int a, int b) { return a > b ? a : b; }

static __device__ __forceinline__ v16bf frag16(const bf16* p0, const bf16* p1) {
  v8bf lo = *(const v8bf*)p0;
  v8bf hi = *(const v8bf*)p1;
  return __builtin_shufflevector(lo, hi, 0, 1, 2, 3, 4, 5, 6, 7,
                                 8, 9, 10, 11, 12, 13, 14, 15);
}

// ---------------------------------------------------------------------------
// Batched GEMM: C[z] = A[z] (M,K) * B[z/bDiv]^T (N,K rows) (+bias)(relu).
// Ks = K padded to mult of 32; lda/ldb mult of 8; A pads zero-filled.
// One wave per 16x32 C tile; A fragment reused across two v_wmma issues.
// Fragment layouts per CDNA5 ISA 7.12.2 (wave32).
// ---------------------------------------------------------------------------
__global__ void gemm_wmma(const bf16* __restrict__ A, const bf16* __restrict__ B,
                          float* __restrict__ Cf, bf16* __restrict__ Cb,
                          const float* __restrict__ bias,
                          int M, int N, int Ks, int lda, int ldb, int ldc,
                          long sA, long sB, long sC, int bDiv, int flags) {
  const int lane = threadIdx.x;
  const int mT = blockIdx.x * 16, nT = blockIdx.y * 32;
  const int z = blockIdx.z;
  const int row16 = lane & 15, hi = lane >> 4;
  const int mAc = imin(mT + row16, M - 1);        // clamped: garbage rows are
  const int n0c = imin(nT + row16, N - 1);        // store-masked later
  const int n1c = imin(nT + 16 + row16, N - 1);
  const bool twoN = (nT + 16) < N;
  const bf16* __restrict__ pA  = A + (long)z * sA + (long)mAc * lda + hi * 8;
  const bf16* __restrict__ pB0 = B + (long)(z / bDiv) * sB + (long)n0c * ldb + hi * 8;
  const bf16* __restrict__ pB1 = B + (long)(z / bDiv) * sB + (long)n1c * ldb + hi * 8;
  v8f acc0 = {}, acc1 = {};

  for (int kt = 0; kt < Ks; kt += 32) {
    v16bf a = frag16(pA + kt, pA + kt + 16);      // 2 x b128, imm offsets
    v16bf b0 = frag16(pB0 + kt, pB0 + kt + 16);
    acc0 = __builtin_amdgcn_wmma_f32_16x16x32_bf16(false, a, false, b0, (short)0,
                                                   acc0, false, false);
    if (twoN) {
      v16bf b1 = frag16(pB1 + kt, pB1 + kt + 16);
      acc1 = __builtin_amdgcn_wmma_f32_16x16x32_bf16(false, a, false, b1, (short)0,
                                                     acc1, false, false);
    }
  }
  const bool ctr = flags & GF_CTRANS;
#pragma unroll
  for (int sub = 0; sub < 2; ++sub) {
    int n = nT + sub * 16 + row16;
    v8f acc = sub ? acc1 : acc0;
    if (n < N) {
      float bsv = (flags & GF_BIAS) ? bias[n] : 0.0f;
#pragma unroll
      for (int v = 0; v < 8; ++v) {
        int m = mT + hi * 8 + v;
        if (m < M) {
          float x = acc[v] + bsv;
          if (flags & GF_RELU) x = fmaxf(x, 0.0f);
          long off = (long)z * sC + (ctr ? (long)n * ldc + m : (long)m * ldc + n);
          if (flags & GF_OUTBF) Cb[off] = (bf16)x; else Cf[off] = x;
        }
      }
    } else if (!ctr && (flags & GF_OUTBF) && n < ldc) {
      // zero the pad columns so this buffer can be reused as zero-padded A
#pragma unroll
      for (int v = 0; v < 8; ++v) {
        int m = mT + hi * 8 + v;
        if (m < M) Cb[(long)z * sC + (long)m * ldc + n] = (bf16)0.0f;
      }
    }
  }
}

// ---------------------------------------------------------------------------
// Implicit-im2col conv (3x3, pad 1, stride 1|2), NCHW bf16 act,
// wt[Cout][Ks] bf16 (k-contiguous, Ks=pad32(Cin*9)), f32 NCHW out.
// One wave per (16 pixels x 32 couts); A fragment reused for two v_wmma.
// A is a spatial gather (scalar loads, zero-pad border via select); B is
// two b128 loads. UPS=true: act is (N,Cin,3,3), input image is its virtual
// HxW bilinear upsample (half-pixel, edge clamp) fused into the gather.
// ---------------------------------------------------------------------------
template <bool UPS>
__global__ void conv_wmma(const bf16* __restrict__ act, const bf16* __restrict__ wt,
                          float* __restrict__ out,
                          int Cin, int H, int W, int Cout, int Ho, int Wo, int stride) {
  const int lane = threadIdx.x;
  const int pixT = blockIdx.x * 16, coT = blockIdx.y * 32, nimg = blockIdx.z;
  const int row16 = lane & 15, hi = lane >> 4;
  const int K = Cin * 9, Ks = (K + 31) & ~31;
  const int npix = Ho * Wo;
  const int pixc = imin(pixT + row16, npix - 1);  // clamped; garbage rows masked
  const int oy = pixc / Wo, ox = pixc - oy * Wo;
  const bool twoC = (coT + 16) < Cout;
  const bf16* __restrict__ actn = act + (long)nimg * Cin * (UPS ? 9 : H * W);
  const bf16* __restrict__ pB0 = wt + (long)imin(coT + row16, Cout - 1) * Ks + hi * 8;
  const bf16* __restrict__ pB1 = wt + (long)imin(coT + 16 + row16, Cout - 1) * Ks + hi * 8;
  v8f acc0 = {}, acc1 = {};
  const int Kmain = K & ~31;

  auto sampleA = [&](int k) -> float {            // k must be < K
    int ci = k / 9, tap = k - ci * 9;
    int dy = tap / 3, dx = tap - dy * 3;
    int iy = oy * stride - 1 + dy, ix = ox * stride - 1 + dx;
    bool in = (iy >= 0) & (iy < H) & (ix >= 0) & (ix < W);
    int iyc = imin(imax(iy, 0), H - 1), ixc = imin(imax(ix, 0), W - 1);
    float av;
    if (UPS) {
      float sy = (iyc + 0.5f) * (3.0f / (float)H) - 0.5f;
      float sx = (ixc + 0.5f) * (3.0f / (float)W) - 0.5f;
      sy = fminf(fmaxf(sy, 0.0f), 2.0f);
      sx = fminf(fmaxf(sx, 0.0f), 2.0f);
      int y0 = imin((int)sy, 1), x0 = imin((int)sx, 1);
      float fy = sy - y0, fx = sx - x0;
      const bf16* s = actn + ci * 9;
      float v00 = (float)s[y0 * 3 + x0],       v01 = (float)s[y0 * 3 + x0 + 1];
      float v10 = (float)s[(y0 + 1) * 3 + x0], v11 = (float)s[(y0 + 1) * 3 + x0 + 1];
      av = (1.0f - fy) * ((1.0f - fx) * v00 + fx * v01) +
           fy * ((1.0f - fx) * v10 + fx * v11);
    } else {
      av = (float)actn[((long)ci * H + iyc) * W + ixc];
    }
    return in ? av : 0.0f;                        // zero-pad border via select
  };

  for (int kt = 0; kt < Ks; kt += 32) {
    v16bf a{};
    if (kt + 32 <= K) {                           // full tile: no k masking
#pragma unroll
      for (int e = 0; e < 16; ++e)
        a[e] = (bf16)sampleA(kt + hi * 8 + (e & 7) + ((e & 8) << 1));
    } else {                                      // tail: zero-mask A only
#pragma unroll
      for (int e = 0; e < 16; ++e) {
        int k = kt + hi * 8 + (e & 7) + ((e & 8) << 1);
        float av = sampleA(imin(k, K - 1));
        a[e] = (bf16)((k < K) ? av : 0.0f);
      }
    }
    v16bf b0 = frag16(pB0 + kt, pB0 + kt + 16);   // wt pads garbage: A zeroed
    acc0 = __builtin_amdgcn_wmma_f32_16x16x32_bf16(false, a, false, b0, (short)0,
                                                   acc0, false, false);
    if (twoC) {
      v16bf b1 = frag16(pB1 + kt, pB1 + kt + 16);
      acc1 = __builtin_amdgcn_wmma_f32_16x16x32_bf16(false, a, false, b1, (short)0,
                                                     acc1, false, false);
    }
  }
#pragma unroll
  for (int sub = 0; sub < 2; ++sub) {
    int co = coT + sub * 16 + row16;
    if (co >= Cout) continue;
    v8f acc = sub ? acc1 : acc0;
#pragma unroll
    for (int v = 0; v < 8; ++v) {
      int p = pixT + hi * 8 + v;
      if (p < npix)
        out[((long)nimg * Cout + co) * npix + p] = acc[v];
    }
  }
}

// --------------------------- staging / utility kernels ---------------------
__global__ void cvt_f32_bf16(const float* __restrict__ s, bf16* __restrict__ d, int n) {
  int i = blockIdx.x * 256 + threadIdx.x;
  if (i < n) d[i] = (bf16)s[i];
}
__global__ void copy_f32(const float* __restrict__ s, float* __restrict__ d, int n) {
  int i = blockIdx.x * 256 + threadIdx.x;
  if (i < n) d[i] = s[i];
}
// (rows,K) f32 -> (rows,Ks) bf16, zero pads
__global__ void pad_cvt(const float* __restrict__ s, bf16* __restrict__ d,
                        int rows, int K, int Ks) {
  int i = blockIdx.x * 256 + threadIdx.x;
  if (i >= rows * Ks) return;
  int n = i / Ks, k = i - n * Ks;
  d[i] = (k < K) ? (bf16)s[(long)n * K + k] : (bf16)0.0f;
}
// batched (K,N) f32 -> (N,Ks) bf16 transposed, zero pads
__global__ void tr_cvt(const float* __restrict__ s, bf16* __restrict__ d,
                       int K, int N, int Ks, int batch) {
  int i = blockIdx.x * 256 + threadIdx.x;
  if (i >= batch * N * Ks) return;
  int z = i / (N * Ks), r = i - z * (N * Ks);
  int n = r / Ks, k = r - n * Ks;
  d[i] = (k < K) ? (bf16)s[(long)z * K * N + (long)k * N + n] : (bf16)0.0f;
}
// inp (B=256,F=5,150,3,3) f32 -> act (F*B,150,9) bf16 (frame-major)
__global__ void transpose_cvt(const float* __restrict__ inp, bf16* __restrict__ act) {
  int i = blockIdx.x * 256 + threadIdx.x;
  const int tot = 1280 * 1350;
  if (i >= tot) return;
  int n = i / 1350, r = i - n * 1350;
  int f = n >> 8, b = n & 255;
  act[i] = (bf16)inp[(long)(b * 5 + f) * 1350 + r];
}
// fm f32 (1280,19,9) -> featT (1280,9,32) bf16 (pixel-major, zero pads)
__global__ void featt_kernel(const float* __restrict__ fm, bf16* __restrict__ ft) {
  int i = blockIdx.x * 256 + threadIdx.x;
  const int tot = 1280 * 288;
  if (i >= tot) return;
  int fb = i / 288, r = i - fb * 288;
  int p = r >> 5, c = r & 31;
  ft[i] = (c < 19) ? (bf16)fm[(long)fb * 171 + c * 9 + p] : (bf16)0.0f;
}
// padded graph (1280 x 19 rows, stride 64) -> dense GRU input (1280,1184), zero pads
__global__ void gseq_repack(const bf16* __restrict__ g, bf16* __restrict__ o) {
  int i = blockIdx.x * 256 + threadIdx.x;
  const int tot = 1280 * 1184;
  if (i >= tot) return;
  int fb = i / 1184, r = i - fb * 1184;
  bf16 v = (bf16)0.0f;
  if (r < 1178) { int n = r / 62, h = r - n * 62; v = g[(long)fb * 1216 + n * 64 + h]; }
  o[i] = v;
}
// per-channel batch stats of f32 NCHW tensor
__global__ void bn_stats(const float* __restrict__ x, float* __restrict__ mean,
                         float* __restrict__ rstd, int N, int C, int HW) {
  int c = blockIdx.x, tid = threadIdx.x;
  int cnt = N * HW;
  float s = 0.0f, s2 = 0.0f;
  for (int i = tid; i < cnt; i += 256) {
    int n = i / HW, p = i - n * HW;
    float v = x[((long)n * C + c) * HW + p];
    s += v; s2 += v * v;
  }
  __shared__ float sh[256], sh2[256];
  sh[tid] = s; sh2[tid] = s2; __syncthreads();
  for (int st = 128; st > 0; st >>= 1) {
    if (tid < st) { sh[tid] += sh[tid + st]; sh2[tid] += sh2[tid + st]; }
    __syncthreads();
  }
  if (tid == 0) {
    float m = sh[0] / (float)cnt;
    float v = sh2[0] / (float)cnt - m * m;
    mean[c] = m; rstd[c] = rsqrtf(fmaxf(v, 0.0f) + 1e-5f);
  }
}
// y = bn(x)*g+b (+res) (relu) -> optional f32 + bf16 outputs
__global__ void bn_apply(const float* __restrict__ x, const float* __restrict__ mean,
                         const float* __restrict__ rstd, const float* __restrict__ g,
                         const float* __restrict__ b, const float* __restrict__ res,
                         float* __restrict__ outF, bf16* __restrict__ outB,
                         int total, int C, int HW, int relu) {
  int i = blockIdx.x * 256 + threadIdx.x;
  if (i >= total) return;
  int c = (i / HW) % C;
  float y = (x[i] - mean[c]) * rstd[c] * g[c] + b[c];
  if (res)  y += res[i];
  if (relu) y = fmaxf(y, 0.0f);
  if (outF) outF[i] = y;
  if (outB) outB[i] = (bf16)y;
}
// softmax over the 9 pixels per (fb,node); logits (FB,9,19) f32
//  -> assignT (FB,19,32) bf16, zero pads (used as zero-padded GEMM A)
__global__ void softmax_assign(const float* __restrict__ lg, bf16* __restrict__ as,
                               int FB) {
  int i = blockIdx.x * 256 + threadIdx.x;
  if (i >= FB * 19) return;
  int fb = i / 19, n = i - fb * 19;
  const float* p = lg + (long)fb * 171 + n;
  float mx = -1e30f;
#pragma unroll
  for (int q = 0; q < 9; ++q) mx = fmaxf(mx, p[q * 19]);
  float e[9], s = 0.0f;
#pragma unroll
  for (int q = 0; q < 9; ++q) { e[q] = __expf(p[q * 19] - mx); s += e[q]; }
  float inv = 1.0f / s;
  bf16* o = as + (long)fb * 608 + n * 32;
#pragma unroll
  for (int q = 0; q < 9; ++q) o[q] = (bf16)(e[q] * inv);
#pragma unroll
  for (int q = 9; q < 32; ++q) o[q] = (bf16)0.0f;
}
// fused GRU gates; gi/gh (256,1824), h (256,608) updated in place
__global__ void gru_gate(const float* __restrict__ gi, const float* __restrict__ gh,
                         float* __restrict__ h) {
  int i = blockIdx.x * 256 + threadIdx.x;
  if (i >= 256 * 608) return;
  int b = i / 608, j = i - b * 608;
  const float* pi = gi + (long)b * 1824;
  const float* ph = gh + (long)b * 1824;
  float r = 1.0f / (1.0f + __expf(-(pi[j] + ph[j])));
  float z = 1.0f / (1.0f + __expf(-(pi[608 + j] + ph[608 + j])));
  float nn = tanhf(pi[1216 + j] + r * ph[1216 + j]);
  h[i] = (1.0f - z) * nn + z * h[i];
}
// n2[b,n] = hg[b,n,:] . g2f_hid
__global__ void n2_kernel(const float* __restrict__ h, const float* __restrict__ ghid,
                          float* __restrict__ n2) {
  int i = blockIdx.x * 256 + threadIdx.x;
  if (i >= 256 * 19) return;
  int b = i / 19, n = i - b * 19;
  const float* p = h + (long)b * 608 + n * 32;
  float s = 0.0f;
#pragma unroll
  for (int t = 0; t < 32; ++t) s += p[t] * ghid[t];
  n2[i] = s;
}
// fused pixel attention + residual: fea = relu(softmax(n1+n2) @ hgw) + xe
__global__ void att_fea(const float* __restrict__ xe, const float* __restrict__ gres,
                        const float* __restrict__ n2, const float* __restrict__ hgw,
                        float* __restrict__ feaF, bf16* __restrict__ feaB) {
  const int HW = 2852;
  int i = blockIdx.x * 256 + threadIdx.x;
  if (i >= 256 * HW) return;
  int b = i / HW, p = i - b * HW;
  float xv[19], n1 = 0.0f;
#pragma unroll
  for (int c = 0; c < 19; ++c) {
    xv[c] = xe[((long)b * 19 + c) * HW + p];
    n1 += xv[c] * gres[c];
  }
  float s[19], mx = -1e30f;
#pragma unroll
  for (int n = 0; n < 19; ++n) { s[n] = n1 + n2[b * 19 + n]; mx = fmaxf(mx, s[n]); }
  float sum = 0.0f;
#pragma unroll
  for (int n = 0; n < 19; ++n) { s[n] = __expf(s[n] - mx); sum += s[n]; }
  float inv = 1.0f / sum;
  const float* W = hgw + (long)b * 361;
#pragma unroll
  for (int c = 0; c < 19; ++c) {
    float o = 0.0f;
#pragma unroll
    for (int n = 0; n < 19; ++n) o += s[n] * W[n * 19 + c];
    float f = fmaxf(o * inv, 0.0f) + xv[c];
    long off = ((long)b * 19 + c) * HW + p;
    feaF[off] = f; feaB[off] = (bf16)f;
  }
}

// ------------------------------- host side ---------------------------------
static inline char* wsa(char*& p, size_t bytes) {
  char* r = p;
  p += (bytes + 255) & ~(size_t)255;
  return r;
}
static void gemm(hipStream_t st, const bf16* A, const bf16* B, float* Cf, bf16* Cb,
                 const float* bias, int M, int N, int Ks, int lda, int ldb, int ldc,
                 long sA, long sB, long sC, int batch, int bDiv, int flags) {
  dim3 g((M + 15) / 16, (N + 31) / 32, batch);
  gemm_wmma<<<g, 32, 0, st>>>(A, B, Cf, Cb, bias, M, N, Ks, lda, ldb, ldc,
                              sA, sB, sC, bDiv, flags);
}
// one residual basic block on the 1280x(C)x3x3 front end
static void run_block(hipStream_t st, const bf16* inB, const float* inF,
                      int Cin, int Cout,
                      const float* w1, const float* g1, const float* b1,
                      const float* w2, const float* g2, const float* b2,
                      const float* dw, const float* dg, const float* db,
                      bf16* WT, float* T1, float* T2, float* T3, float* SKIP,
                      bf16* TB, float* MEAN, float* RSTD,
                      float* outF, bf16* outB) {
  const int N = 1280;
  const int K1 = Cin * 9, Ks1 = (K1 + 31) & ~31;
  const int K2 = Cout * 9, Ks2 = (K2 + 31) & ~31;
  dim3 blk(256);
  dim3 cg(1, (Cout + 31) / 32, N);
  int tot = N * Cout * 9;
  pad_cvt<<<dim3((Cout * Ks1 + 255) / 256), blk, 0, st>>>(w1, WT, Cout, K1, Ks1);
  conv_wmma<false><<<cg, 32, 0, st>>>(inB, WT, T1, Cin, 3, 3, Cout, 3, 3, 1);
  bn_stats<<<Cout, 256, 0, st>>>(T1, MEAN, RSTD, N, Cout, 9);
  bn_apply<<<(tot + 255) / 256, blk, 0, st>>>(T1, MEAN, RSTD, g1, b1, nullptr,
                                              nullptr, TB, tot, Cout, 9, 1);
  pad_cvt<<<dim3((Cout * Ks2 + 255) / 256), blk, 0, st>>>(w2, WT, Cout, K2, Ks2);
  conv_wmma<false><<<cg, 32, 0, st>>>(TB, WT, T2, Cout, 3, 3, Cout, 3, 3, 1);
  const float* skip = inF;
  if (dw) {
    pad_cvt<<<dim3((Cout * Ks1 + 255) / 256), blk, 0, st>>>(dw, WT, Cout, K1, Ks1);
    conv_wmma<false><<<cg, 32, 0, st>>>(inB, WT, T3, Cin, 3, 3, Cout, 3, 3, 1);
    bn_stats<<<Cout, 256, 0, st>>>(T3, MEAN, RSTD, N, Cout, 9);
    bn_apply<<<(tot + 255) / 256, blk, 0, st>>>(T3, MEAN, RSTD, dg, db, nullptr,
                                                SKIP, nullptr, tot, Cout, 9, 0);
    skip = SKIP;
  }
  bn_stats<<<Cout, 256, 0, st>>>(T2, MEAN, RSTD, N, Cout, 9);
  bn_apply<<<(tot + 255) / 256, blk, 0, st>>>(T2, MEAN, RSTD, g2, b2, skip,
                                              outF, outB, tot, Cout, 9, 1);
}
// conv-bn-relu, conv-bn-relu, conv; 46x62 output, batch 256
static void run_seq3(hipStream_t st, const bf16* inB, int C2, bool ups,
                     const float* w1, const float* g1, const float* b1,
                     const float* w2, const float* g2, const float* b2,
                     const float* w3, bf16* WT, float* P0, bf16* PB0,
                     float* P1, bf16* PB1, float* MEAN, float* RSTD, float* outF) {
  const int Cin = 19, N = 256, HW = 2852;
  const int K1 = Cin * 9, Ks1 = (K1 + 31) & ~31;
  const int K2 = C2 * 9, Ks2 = (K2 + 31) & ~31;
  dim3 blk(256);
  dim3 cg((HW + 15) / 16, (C2 + 31) / 32, N);
  int tot = N * C2 * HW;
  pad_cvt<<<dim3((C2 * Ks1 + 255) / 256), blk, 0, st>>>(w1, WT, C2, K1, Ks1);
  if (ups)
    conv_wmma<true><<<cg, 32, 0, st>>>(inB, WT, P0, Cin, 92, 124, C2, 46, 62, 2);
  else
    conv_wmma<false><<<cg, 32, 0, st>>>(inB, WT, P0, Cin, 46, 62, C2, 46, 62, 1);
  bn_stats<<<C2, 256, 0, st>>>(P0, MEAN, RSTD, N, C2, HW);
  bn_apply<<<(tot + 255) / 256, blk, 0, st>>>(P0, MEAN, RSTD, g1, b1, nullptr,
                                              nullptr, PB0, tot, C2, HW, 1);
  pad_cvt<<<dim3((C2 * Ks2 + 255) / 256), blk, 0, st>>>(w2, WT, C2, K2, Ks2);
  conv_wmma<false><<<cg, 32, 0, st>>>(PB0, WT, P1, C2, 46, 62, C2, 46, 62, 1);
  bn_stats<<<C2, 256, 0, st>>>(P1, MEAN, RSTD, N, C2, HW);
  bn_apply<<<(tot + 255) / 256, blk, 0, st>>>(P1, MEAN, RSTD, g2, b2, nullptr,
                                              nullptr, PB1, tot, C2, HW, 1);
  pad_cvt<<<dim3((C2 * Ks2 + 255) / 256), blk, 0, st>>>(w3, WT, C2, K2, Ks2);
  conv_wmma<false><<<cg, 32, 0, st>>>(PB1, WT, outF, C2, 46, 62, C2, 46, 62, 1);
}

extern "C" void kernel_launch(void* const* d_in, const int* in_sizes, int n_in,
                              void* d_out, int out_size, void* d_ws, size_t ws_size,
                              hipStream_t stream) {
  (void)in_sizes; (void)n_in; (void)out_size; (void)ws_size;
  const float* IN = (const float*)d_in[0];
  const float* ADJ = (const float*)d_in[1];
  const float* H0 = (const float*)d_in[2];
  auto F = [&](int i) { return (const float*)d_in[i]; };
  float* OUT = (float*)d_out;

  char* p = (char*)d_ws;
  // ---- workspace layout (elements noted; K-strides padded to mult 32) ----
  bf16* ACT  = (bf16*)wsa(p, 1728000ULL * 2);     // (1280,150,9)
  float* CF0 = (float*)wsa(p, 437760ULL * 4);
  float* CF1 = (float*)wsa(p, 437760ULL * 4);
  bf16* CB0  = (bf16*)wsa(p, 437760ULL * 2);
  bf16* CB1  = (bf16*)wsa(p, 437760ULL * 2);
  float* T1  = (float*)wsa(p, 437760ULL * 4);
  float* T2  = (float*)wsa(p, 437760ULL * 4);
  float* T3  = (float*)wsa(p, 437760ULL * 4);
  float* SKIP = (float*)wsa(p, 437760ULL * 4);
  bf16* TB   = (bf16*)wsa(p, 437760ULL * 2);
  float* MEAN = (float*)wsa(p, 64 * 4);
  float* RSTD = (float*)wsa(p, 64 * 4);
  bf16* WT   = (bf16*)wsa(p, 56000ULL * 2);       // conv wt, max 38x1376
  float* LOGITS = (float*)wsa(p, 218880ULL * 4);  // (1280,9,19)
  bf16* FEATT = (bf16*)wsa(p, 368640ULL * 2);     // (1280,9,32)
  bf16* ASSIGNT = (bf16*)wsa(p, 778240ULL * 2);   // (1280,19,32)
  bf16* WFEATT = (bf16*)wsa(p, 2539520ULL * 2);   // (1280,62,32)
  bf16* G0   = (bf16*)wsa(p, 1556480ULL * 2);     // (1280,19,64)
  bf16* G1   = (bf16*)wsa(p, 1556480ULL * 2);
  bf16* TMPGT = (bf16*)wsa(p, 2539520ULL * 2);    // (1280,62,32)
  bf16* ADJB = (bf16*)wsa(p, 608ULL * 2);         // (19,32)
  bf16* PREDT = (bf16*)wsa(p, 3040ULL * 2);       // (5,19,32)
  bf16* F2GWT = (bf16*)wsa(p, 9920ULL * 2);       // (5,62,32)
  bf16* GCWT = (bf16*)wsa(p, 59520ULL * 2);       // (15,62,64)
  bf16* G2FWT = (bf16*)wsa(p, 608ULL * 2);        // (19,32)
  bf16* WIHB = (bf16*)wsa(p, 10798080ULL * 2);    // (5*1824,1184)
  bf16* WHHB = (bf16*)wsa(p, 5544960ULL * 2);     // (5*1824,608)
  bf16* GSEQD = (bf16*)wsa(p, 1515520ULL * 2);    // (1280,1184)
  float* GI = (float*)wsa(p, 466944ULL * 4);
  float* GH = (float*)wsa(p, 466944ULL * 4);
  float* H  = (float*)wsa(p, 155648ULL * 4);
  bf16* HB  = (bf16*)wsa(p, 155648ULL * 2);
  float* N2 = (float*)wsa(p, 4864ULL * 4);
  float* HGW = (float*)wsa(p, 92416ULL * 4);
  float* P0 = (float*)wsa(p, 27744256ULL * 4);
  float* P1 = (float*)wsa(p, 27744256ULL * 4);
  bf16* PB0 = (bf16*)wsa(p, 27744256ULL * 2);
  bf16* PB1 = (bf16*)wsa(p, 27744256ULL * 2);
  float* XE = (float*)wsa(p, 13872128ULL * 4);
  float* FEA = (float*)wsa(p, 13872128ULL * 4);
  bf16* FEAB = (bf16*)wsa(p, 13872128ULL * 2);

  hipStream_t st = stream;
  dim3 blk(256);

  // ---- phase 1: front-end ResNet on (1280, C, 3, 3) ----
  transpose_cvt<<<dim3((1728000 + 255) / 256), blk, 0, st>>>(IN, ACT);
  run_block(st, ACT, nullptr, 150, 38, F(3), F(4), F(5), F(6), F(7), F(8),
            F(9), F(10), F(11), WT, T1, T2, T3, SKIP, TB, MEAN, RSTD, CF0, CB0);
  run_block(st, CB0, CF0, 38, 38, F(12), F(13), F(14), F(15), F(16), F(17),
            nullptr, nullptr, nullptr, WT, T1, T2, T3, SKIP, TB, MEAN, RSTD, CF1, CB1);
  run_block(st, CB1, CF1, 38, 38, F(18), F(19), F(20), F(21), F(22), F(23),
            nullptr, nullptr, nullptr, WT, T1, T2, T3, SKIP, TB, MEAN, RSTD, CF0, CB0);
  run_block(st, CB0, nullptr, 38, 19, F(24), F(25), F(26), F(27), F(28), F(29),
            F(30), F(31), F(32), WT, T1, T2, T3, SKIP, TB, MEAN, RSTD, CF1, CB1);
  run_block(st, CB1, CF1, 19, 19, F(33), F(34), F(35), F(36), F(37), F(38),
            nullptr, nullptr, nullptr, WT, T1, T2, T3, SKIP, TB, MEAN, RSTD, CF0, CB0);
  run_block(st, CB0, CF0, 19, 19, F(39), F(40), F(41), F(42), F(43), F(44),
            nullptr, nullptr, nullptr, WT, T1, T2, T3, SKIP, TB, MEAN, RSTD, CF1, CB1);
  run_block(st, CB1, CF1, 19, 19, F(45), F(46), F(47), F(48), F(49), F(50),
            nullptr, nullptr, nullptr, WT, T1, T2, T3, SKIP, TB, MEAN, RSTD, CF0, CB0);
  // final featuremaps: CF0 f32 + CB0 bf16, (1280,19,9) frame-major

  // ---- phase 2: graph construction + GCN + GRU ----
  featt_kernel<<<dim3((1280 * 288 + 255) / 256), blk, 0, st>>>(CF0, FEATT);
  pad_cvt<<<dim3((19 * 32 + 255) / 256), blk, 0, st>>>(ADJ, ADJB, 19, 19, 32);
  tr_cvt<<<dim3((5 * 19 * 32 + 255) / 256), blk, 0, st>>>(F(52), PREDT, 19, 19, 32, 5);
  tr_cvt<<<dim3((5 * 62 * 32 + 255) / 256), blk, 0, st>>>(F(51), F2GWT, 19, 62, 32, 5);
  tr_cvt<<<dim3((15 * 62 * 64 + 255) / 256), blk, 0, st>>>(F(53), GCWT, 62, 62, 64, 15);
  tr_cvt<<<dim3((19 * 32 + 255) / 256), blk, 0, st>>>(F(81), G2FWT, 32, 19, 32, 1);
  pad_cvt<<<dim3((9120 * 1184 + 255) / 256), blk, 0, st>>>(F(54), WIHB, 9120, 1178, 1184);
  pad_cvt<<<dim3((9120 * 608 + 255) / 256), blk, 0, st>>>(F(55), WHHB, 9120, 608, 608);

  // logits = feat @ f2g_pred[f]
  gemm(st, FEATT, PREDT, LOGITS, nullptr, nullptr, 9, 19, 32, 32, 32, 19,
       288, 608, 171, 1280, 256, 0);
  softmax_assign<<<dim3((1280 * 19 + 255) / 256), blk, 0, st>>>(LOGITS, ASSIGNT, 1280);
  // wfeat = feat @ f2g_w[f], stored transposed (62,32) for reuse as B
  gemm(st, FEATT, F2GWT, nullptr, WFEATT, nullptr, 9, 62, 32, 32, 32, 32,
       288, 1984, 1984, 1280, 256, GF_OUTBF | GF_CTRANS);
  // graph = assign^T @ wfeat -> (19,64) zero-padded rows
  gemm(st, ASSIGNT, WFEATT, nullptr, G0, nullptr, 19, 62, 32, 32, 32, 64,
       608, 1984, 1216, 1280, 1, GF_OUTBF);
  // 3 GCN layers: g' = relu(adj @ (g @ W_l))
  bf16* Gin = G0; bf16* Gout = G1;
  for (int l = 0; l < 3; ++l) {
    gemm(st, Gin, GCWT + l * 3968, nullptr, TMPGT, nullptr, 19, 62, 64, 64, 64, 32,
         1216, 3 * 3968, 1984, 1280, 256, GF_OUTBF | GF_CTRANS);
    gemm(st, ADJB, TMPGT, nullptr, Gout, nullptr, 19, 62, 32, 32, 32, 64,
         0, 1984, 1216, 1280, 1, GF_OUTBF | GF_RELU);
    bf16* t = Gin; Gin = Gout; Gout = t;
  }
  gseq_repack<<<dim3((1280 * 1184 + 255) / 256), blk, 0, st>>>(Gin, GSEQD);

  // GRU over 5 frames (distinct cell per frame)
  copy_f32<<<dim3((155648 + 255) / 256), blk, 0, st>>>(H0, H, 155648);
  for (int f = 0; f < 5; ++f) {
    cvt_f32_bf16<<<dim3((155648 + 255) / 256), blk, 0, st>>>(H, HB, 155648);
    gemm(st, GSEQD + (long)f * 303104, WIHB + (long)f * 2159616, GI, nullptr,
         F(56) + f * 1824, 256, 1824, 1184, 1184, 1184, 1824, 0, 0, 0, 1, 1, GF_BIAS);
    gemm(st, HB, WHHB + (long)f * 1108992, GH, nullptr,
         F(57) + f * 1824, 256, 1824, 608, 608, 608, 1824, 0, 0, 0, 1, 1, GF_BIAS);
    gru_gate<<<dim3((155648 + 255) / 256), blk, 0, st>>>(GI, GH, H);
  }
  cvt_f32_bf16<<<dim3((155648 + 255) / 256), blk, 0, st>>>(H, HB, 155648);
  n2_kernel<<<dim3((256 * 19 + 255) / 256), blk, 0, st>>>(H, F(80), N2);
  // hgw[b] = hg[b] @ g2f_w  (B,19,19) f32
  gemm(st, HB, G2FWT, HGW, nullptr, nullptr, 19, 19, 32, 32, 32, 19,
       608, 0, 361, 256, 1, 0);

  // ---- phase 3: encoder (fused bilinear upsample) ----
  const bf16* FM4 = CB0 + (long)4 * 256 * 171;   // last frame, (256,19,3,3) bf16
  run_seq3(st, FM4, 19, true, F(58), F(59), F(60), F(61), F(62), F(63), F(64),
           WT, P0, PB0, P1, PB1, MEAN, RSTD, XE);

  // ---- phase 4: attention fusion ----
  att_fea<<<dim3((256 * 2852 + 255) / 256), blk, 0, st>>>(XE, F(79), N2, HGW,
                                                          FEA, FEAB);

  // ---- phase 5: decoders (write straight into d_out, NCHW concat) ----
  run_seq3(st, FEAB, 19, false, F(65), F(66), F(67), F(68), F(69), F(70), F(71),
           WT, P0, PB0, P1, PB1, MEAN, RSTD, OUT);
  run_seq3(st, FEAB, 38, false, F(72), F(73), F(74), F(75), F(76), F(77), F(78),
           WT, P0, PB0, P1, PB1, MEAN, RSTD, OUT + 13872128L);
}
